// ODEBlock_22488448762447
// MI455X (gfx1250) — compile-verified, matrix operand order checked by reading, not run on hardware
//
#include <hip/hip_runtime.h>

// ---------------------------------------------------------------------------
// Neural ODE (sonode-style) on MI455X / gfx1250.
//   dy/dt = tanh(y@W1 + b1)@W2 + b2,  t: 0 -> 1, then per-row column gather.
// Fixed-step RK4 (16 steps): deterministic, graph-capture safe, LTE ~ dt^5
// << tol=1e-3 for this smooth bounded field.
//
// Compute: f16 WMMA (v_wmma_f32_16x16x32_f16), f32 accumulation.
// Data movement: weights pre-packed into the exact WMMA B-fragment layout;
// each block stages its B panel into LDS once via the gfx1250 async-to-LDS
// path (global_load_async_to_lds_b128 + s_wait_asynccnt), then 8 waves with
// disjoint 16x64 row strips consume it via ds_load_b128. Weights are
// L2-resident (192 MB L2) for the whole run; HBM traffic = state tensors.
// ---------------------------------------------------------------------------

typedef __attribute__((ext_vector_type(16))) _Float16 v16h;
typedef __attribute__((ext_vector_type(8)))  float    v8f;
typedef __attribute__((ext_vector_type(4)))  int      v4i;

#define B_   4096
#define D_   256
#define H_   1024
#define K_   128
#define NSTEPS 16

#if __has_builtin(__builtin_amdgcn_global_load_async_to_lds_b128)
#define HAVE_ASYNC_LDS 1
typedef __attribute__((address_space(1))) v4i* gv4i_p;   // global 16B vector
typedef __attribute__((address_space(3))) v4i* lv4i_p;   // LDS 16B vector
#else
#define HAVE_ASYNC_LDS 0
#endif

// ---- stage a B weight panel (KT*4 fragments of 512 halves) into LDS.
// Panel fragment p (= kt*4 + j) comes from global fragment (p>>2)*NT +
// nGroup*4 + (p&3). Copy granularity: 16B chunks (64 per fragment).
template <int KT, int NT>
__device__ inline void stage_b_panel(const _Float16* __restrict__ Wp,
                                     int nGroup, _Float16* smem) {
  const int chunks = KT * 4 * 64;
  for (int c = threadIdx.x; c < chunks; c += blockDim.x) {
    int p = c >> 6;        // panel fragment
    int w = c & 63;        // 16B chunk within fragment
    int gf = (p >> 2) * NT + nGroup * 4 + (p & 3);
    const _Float16* src = Wp + ((size_t)gf * 512 + w * 8);
    _Float16*       dst = smem + ((size_t)p * 512 + w * 8);
#if HAVE_ASYNC_LDS
    __builtin_amdgcn_global_load_async_to_lds_b128((gv4i_p)src, (lv4i_p)dst,
                                                   0, 0);
#else
    *(uint4*)dst = *(const uint4*)src;
#endif
  }
#if HAVE_ASYNC_LDS
#if __has_builtin(__builtin_amdgcn_s_wait_asynccnt)
  __builtin_amdgcn_s_wait_asynccnt(0);
#else
  asm volatile("s_wait_asynccnt 0" ::: "memory");
#endif
#endif
  __syncthreads();
}

// ---- pack f32 row-major [Kdim x Ndim] weight into WMMA f16 B-fragment layout.
// Fragment tile = 32(K) x 16(N): lane L covers column (L&15), K base
// (L>>4)*16, holding 16 consecutive K values contiguously.
__global__ void pack_w_kernel(const float* __restrict__ W,
                              _Float16* __restrict__ Wp,
                              int Kdim, int Ndim) {
  int i = blockIdx.x * blockDim.x + threadIdx.x;
  int total = Kdim * Ndim;
  if (i >= total) return;
  int t    = i & 15;
  int lane = (i >> 4) & 31;
  int tile = i >> 9;          // 512 elems per fragment
  int NT   = Ndim >> 4;
  int nt   = tile % NT;
  int kt   = tile / NT;
  int k = kt * 32 + ((lane >> 4) << 4) + t;
  int n = nt * 16 + (lane & 15);
  Wp[i] = (_Float16)W[(size_t)k * Ndim + n];
}

// ---- y = x (f32 master state), ytmp = (f16)x (A operand of first f-eval)
__global__ void init_state_kernel(const float* __restrict__ x,
                                  float* __restrict__ y,
                                  _Float16* __restrict__ ytmp, int n) {
  int i = blockIdx.x * blockDim.x + threadIdx.x;
  if (i < n) { float v = x[i]; y[i] = v; ytmp[i] = (_Float16)v; }
}

// ---- GEMM1: h = tanh(ytmp @ W1 + b1), [B x D]@[D x H] -> f16 [B x H].
// Block: 8 waves, 128 rows x 64 cols; wave: 16 x 64 strip (4 accumulators).
__global__ __launch_bounds__(256) void gemm1_tanh_kernel(
    const _Float16* __restrict__ ytmp, const _Float16* __restrict__ W1p,
    const float* __restrict__ b1, _Float16* __restrict__ h) {
  extern __shared__ _Float16 smem[];          // 8*4*512 halves = 32 KiB
  const int KT = D_ / 32;                     // 8 K-iterations
  const int NT = H_ >> 4;                     // 64 N fragment tiles
  int nGroup = blockIdx.x % (H_ / 64);
  int mBlock = blockIdx.x / (H_ / 64);

  stage_b_panel<KT, NT>(W1p, nGroup, smem);

  int waveId = threadIdx.x >> 5;
  int lane   = threadIdx.x & 31;
  int rowBase = mBlock * 128 + waveId * 16;
  int arow  = rowBase + (lane & 15);
  int akoff = (lane >> 4) << 4;

  v8f acc0 = {}, acc1 = {}, acc2 = {}, acc3 = {};
  for (int kt = 0; kt < KT; ++kt) {
    v16h a = *(const v16h*)(ytmp + (size_t)arow * D_ + kt * 32 + akoff);
    const _Float16* bp = smem + ((size_t)(kt * 4) * 512 + lane * 16);
    v16h b0 = *(const v16h*)(bp);
    v16h b1v = *(const v16h*)(bp + 512);
    v16h b2v = *(const v16h*)(bp + 1024);
    v16h b3v = *(const v16h*)(bp + 1536);
    acc0 = __builtin_amdgcn_wmma_f32_16x16x32_f16(false, a, false, b0,  (short)0, acc0, false, false);
    acc1 = __builtin_amdgcn_wmma_f32_16x16x32_f16(false, a, false, b1v, (short)0, acc1, false, false);
    acc2 = __builtin_amdgcn_wmma_f32_16x16x32_f16(false, a, false, b2v, (short)0, acc2, false, false);
    acc3 = __builtin_amdgcn_wmma_f32_16x16x32_f16(false, a, false, b3v, (short)0, acc3, false, false);
  }

  // C/D layout: lane holds col (lane&15); VGPR r is row r + 8*(lane>>4).
  int rbase = rowBase + ((lane >> 4) << 3);
  int cbase = nGroup * 64 + (lane & 15);
  v8f accs[4] = {acc0, acc1, acc2, acc3};
#pragma unroll
  for (int j = 0; j < 4; ++j) {
    int col = cbase + j * 16;
    float bb = b1[col];
#pragma unroll
    for (int r = 0; r < 8; ++r) {
      float v = tanhf(accs[j][r] + bb);
      h[(size_t)(rbase + r) * H_ + col] = (_Float16)v;
    }
  }
}

// ---- GEMM2 + fused RK4 stage combine:
//   dy = h @ W2 + b2   [B x H]@[H x D] -> f32 tile (registers)
//   stage 0: acc = dy;        ytmp = f16(y + dt/2*dy)
//   stage 1: acc += 2*dy;     ytmp = f16(y + dt/2*dy)
//   stage 2: acc += 2*dy;     ytmp = f16(y + dt  *dy)
//   stage 3: y += dt/6*(acc+dy);   ytmp = f16(y)
__global__ __launch_bounds__(256) void gemm2_stage_kernel(
    const _Float16* __restrict__ h, const _Float16* __restrict__ W2p,
    const float* __restrict__ b2, float* __restrict__ y,
    float* __restrict__ accbuf, _Float16* __restrict__ ytmp,
    float dt, int stage) {
  extern __shared__ _Float16 smem[];          // 32*4*512 halves = 128 KiB
  const int KT = H_ / 32;                     // 32 K-iterations
  const int NT = D_ >> 4;                     // 16 N fragment tiles
  int nGroup = blockIdx.x % (D_ / 64);
  int mBlock = blockIdx.x / (D_ / 64);

  stage_b_panel<KT, NT>(W2p, nGroup, smem);

  int waveId = threadIdx.x >> 5;
  int lane   = threadIdx.x & 31;
  int rowBase = mBlock * 128 + waveId * 16;
  int arow  = rowBase + (lane & 15);
  int akoff = (lane >> 4) << 4;

  v8f acc0 = {}, acc1 = {}, acc2 = {}, acc3 = {};
  for (int kt = 0; kt < KT; ++kt) {
    v16h a = *(const v16h*)(h + (size_t)arow * H_ + kt * 32 + akoff);
    const _Float16* bp = smem + ((size_t)(kt * 4) * 512 + lane * 16);
    v16h b0 = *(const v16h*)(bp);
    v16h b1v = *(const v16h*)(bp + 512);
    v16h b2v = *(const v16h*)(bp + 1024);
    v16h b3v = *(const v16h*)(bp + 1536);
    acc0 = __builtin_amdgcn_wmma_f32_16x16x32_f16(false, a, false, b0,  (short)0, acc0, false, false);
    acc1 = __builtin_amdgcn_wmma_f32_16x16x32_f16(false, a, false, b1v, (short)0, acc1, false, false);
    acc2 = __builtin_amdgcn_wmma_f32_16x16x32_f16(false, a, false, b2v, (short)0, acc2, false, false);
    acc3 = __builtin_amdgcn_wmma_f32_16x16x32_f16(false, a, false, b3v, (short)0, acc3, false, false);
  }

  int rbase = rowBase + ((lane >> 4) << 3);
  int cbase = nGroup * 64 + (lane & 15);
  v8f accs[4] = {acc0, acc1, acc2, acc3};
#pragma unroll
  for (int j = 0; j < 4; ++j) {
    int col = cbase + j * 16;
    float bb = b2[col];
#pragma unroll
    for (int r = 0; r < 8; ++r) {
      float dy = accs[j][r] + bb;
      size_t off = (size_t)(rbase + r) * D_ + col;
      if (stage == 0) {
        accbuf[off] = dy;
        ytmp[off]   = (_Float16)(y[off] + 0.5f * dt * dy);
      } else if (stage == 1) {
        accbuf[off] += 2.0f * dy;
        ytmp[off]   = (_Float16)(y[off] + 0.5f * dt * dy);
      } else if (stage == 2) {
        accbuf[off] += 2.0f * dy;
        ytmp[off]   = (_Float16)(y[off] + dt * dy);
      } else {
        float yn = y[off] + (dt / 6.0f) * (accbuf[off] + dy);
        y[off]    = yn;
        ytmp[off] = (_Float16)yn;
      }
    }
  }
}

// ---- out[b, j] = y[b, indices[b, j]]
__global__ void gather_kernel(const float* __restrict__ y,
                              const int* __restrict__ idx,
                              float* __restrict__ out, int n) {
  int i = blockIdx.x * blockDim.x + threadIdx.x;
  if (i >= n) return;
  int b = i / K_;
  out[i] = y[(size_t)b * D_ + idx[i]];
}

extern "C" void kernel_launch(void* const* d_in, const int* in_sizes, int n_in,
                              void* d_out, int out_size, void* d_ws, size_t ws_size,
                              hipStream_t stream) {
  (void)in_sizes; (void)n_in; (void)out_size; (void)ws_size;
  const float* x   = (const float*)d_in[0];   // [B, D]
  const float* W1  = (const float*)d_in[1];   // [D, H]
  const float* b1  = (const float*)d_in[2];   // [H]
  const float* W2  = (const float*)d_in[3];   // [H, D]
  const float* b2  = (const float*)d_in[4];   // [D]
  const int*   idx = (const int*)d_in[5];     // [B, K]
  float* out = (float*)d_out;                 // [B, K]

  // Workspace carve-out (~19 MiB total)
  float*     y     = (float*)d_ws;                        // B*D f32
  float*     accb  = y + (size_t)B_ * D_;                 // B*D f32
  _Float16*  ytmp  = (_Float16*)(accb + (size_t)B_ * D_); // B*D f16
  _Float16*  hbuf  = ytmp + (size_t)B_ * D_;              // B*H f16
  _Float16*  W1p   = hbuf + (size_t)B_ * H_;              // D*H f16 packed
  _Float16*  W2p   = W1p + (size_t)D_ * H_;               // H*D f16 packed

  // 1) pack weights into WMMA B-fragment layout (L2-resident thereafter)
  {
    int total = D_ * H_;
    pack_w_kernel<<<(total + 255) / 256, 256, 0, stream>>>(W1, W1p, D_, H_);
    pack_w_kernel<<<(total + 255) / 256, 256, 0, stream>>>(W2, W2p, H_, D_);
  }

  // 2) init state
  {
    int total = B_ * D_;
    init_state_kernel<<<(total + 255) / 256, 256, 0, stream>>>(x, y, ytmp, total);
  }

  // 3) RK4: 16 steps x 4 stages, 2 fused WMMA kernels per stage
  const float dt = 1.0f / (float)NSTEPS;
  const int g1_blocks = (B_ / 128) * (H_ / 64);   // 512 blocks
  const int g2_blocks = (B_ / 128) * (D_ / 64);   // 128 blocks
  const size_t g1_lds = (size_t)(D_ / 32) * 4 * 512 * sizeof(_Float16); // 32 KiB
  const size_t g2_lds = (size_t)(H_ / 32) * 4 * 512 * sizeof(_Float16); // 128 KiB
  for (int step = 0; step < NSTEPS; ++step) {
    for (int stage = 0; stage < 4; ++stage) {
      gemm1_tanh_kernel<<<g1_blocks, 256, g1_lds, stream>>>(ytmp, W1p, b1, hbuf);
      gemm2_stage_kernel<<<g2_blocks, 256, g2_lds, stream>>>(hbuf, W2p, b2, y,
                                                             accb, ytmp, dt, stage);
    }
  }

  // 4) final gather: out[b, j] = y[b, indices[b, j]]
  {
    int total = B_ * K_;
    gather_kernel<<<(total + 255) / 256, 256, 0, stream>>>(y, idx, out, total);
  }
}
